// AttentionModel_8203387535686
// MI455X (gfx1250) — matrix operand order repaired and checked by reading, not compiled, not run on hardware
//
#include <hip/hip_runtime.h>

// ---------------------------------------------------------------------------
// Problem constants (from reference): S=512, B=64, I=128, H=512, O=1
// ---------------------------------------------------------------------------
#define S_LEN 512
#define BATCH 64
#define IN_F  128
#define HID   512

typedef __attribute__((ext_vector_type(16))) __bf16 v16bf;
typedef __attribute__((ext_vector_type(8)))  float  v8f;
typedef unsigned int v4u __attribute__((ext_vector_type(4)));
typedef int v8i_t __attribute__((ext_vector_type(8)));
typedef int v4i_t __attribute__((ext_vector_type(4)));

__device__ __forceinline__ unsigned short f2bf(float f) {
    __bf16 h = (__bf16)f;                       // RNE convert
    return __builtin_bit_cast(unsigned short, h);
}
__device__ __forceinline__ float bf2f(unsigned short u) {
    __bf16 h = __builtin_bit_cast(__bf16, u);
    return (float)h;
}

// A-fragment (16x32 bf16, MxK) gather.  Per CDNA5 ISA 7.12.2:
//   lane l: M = l % 16
//   lanes 0-15 : halves 0..7 -> K = 0..7,   halves 8..15 -> K = 16..23
//   lanes 16-31: halves 0..7 -> K = 8..15,  halves 8..15 -> K = 24..31
// Each half-run is 8 contiguous bf16 (16 B, 16 B-aligned) -> two b128 loads.
__device__ __forceinline__ v16bf gather_a_u16(const unsigned short* rowptr, int kbase) {
    union { uint4 q[2]; v16bf v; } u;
    u.q[0] = *(const uint4*)(rowptr + kbase);
    u.q[1] = *(const uint4*)(rowptr + kbase + 16);
    return u.v;
}

__device__ __forceinline__ v16bf gather_a_f32(const float* rowptr, int kbase) {
    v16bf a;
#pragma unroll
    for (int j = 0; j < 8; ++j)
        a[j] = (__bf16)rowptr[kbase + j];
#pragma unroll
    for (int j = 0; j < 8; ++j)
        a[8 + j] = (__bf16)rowptr[kbase + 16 + j];
    return a;
}

// B-fragment load from a pre-swizzled weight buffer.  Tile (kt, nt): each lane
// holds 16 contiguous bf16 (32B) -> a single vector load.
__device__ __forceinline__ v16bf load_b(const unsigned short* wp, int nt, int kt,
                                        int KT, int lane) {
    const v16bf* p = (const v16bf*)wp;
    return p[((nt * KT + kt) * 32 + lane)];
}

__device__ __forceinline__ v8f wmma_bf16(v16bf a, v16bf b, v8f c) {
    return __builtin_amdgcn_wmma_f32_16x16x32_bf16(
        false, a, false, b, (short)0, c, false, false);
}

// ---------------------------------------------------------------------------
// TDM: 1-D contiguous fp32 copy global -> LDS via tensor_load_to_lds.
// D# per CDNA5 ISA:
//   group0: [1:0]=count=1 | lds_addr | global_addr[56:0] | type=2
//   group1: data_size=2 (4B), tensor_dim0 = tile_dim0 = nelem, tensor_dim1=1
//   groups 2/3 (+extra group on this toolchain): zero (<=2D tensor)
// 6-arg builtin variant: (v4u, v8i, v4i, v4i, v8i, i32 cpol).
// Issued by one wave; completion via s_wait_tensorcnt + workgroup barrier.
// ---------------------------------------------------------------------------
__device__ __forceinline__ void tdm_load_1d_f32(unsigned lds_off,
                                                const float* gptr,
                                                unsigned nelem) {
    unsigned long long ga = (unsigned long long)(uintptr_t)gptr;
    v4u g0;
    g0[0] = 1u;                                            // count = 1 (valid)
    g0[1] = lds_off;                                       // lds_addr (bytes)
    g0[2] = (unsigned)ga;                                  // global_addr[31:0]
    g0[3] = (unsigned)((ga >> 32) & 0x1FFFFFFu) | (2u << 30); // [56:32]|type=2
    v8i_t g1;
    g1[0] = 0x00020000;                                    // data_size = 4B
    g1[1] = (int)((nelem & 0xFFFFu) << 16);                // tensor_dim0 lo
    g1[2] = (int)(((nelem >> 16) & 0xFFFFu) | (1u << 16)); // dim0 hi | dim1=1
    g1[3] = (int)((nelem & 0xFFFFu) << 16);                // tile_dim0
    g1[4] = 0; g1[5] = 0; g1[6] = 0; g1[7] = 0;
    v4i_t gz = {0, 0, 0, 0};
    v8i_t gz8 = {0, 0, 0, 0, 0, 0, 0, 0};
    __builtin_amdgcn_tensor_load_to_lds(g0, g1, gz, gz, gz8, 0);
}

// ---------------------------------------------------------------------------
// Kernel 0: pack weight matrix W (N rows x K cols, fp32) into bf16
// B-fragment layout for D = A x B with B[k][n] = W[n][k].
// ---------------------------------------------------------------------------
__global__ void prep_b_kernel(const float* __restrict__ W,
                              unsigned short* __restrict__ out,
                              int K, int N) {
    int e = blockIdx.x * blockDim.x + threadIdx.x;
    if (e >= K * N) return;
    int n = e / K, k = e % K;
    int kt = k >> 5, kk = k & 31;
    int nt = n >> 4, nn = n & 15;
    int lane = nn + (((kk >> 3) & 1) ? 16 : 0);
    int j = (kk & 7) + ((kk >= 16) ? 8 : 0);
    int KT = K >> 5;
    out[(((nt * KT + kt) * 32 + lane) << 4) + j] = f2bf(W[n * K + k]);
}

// ---------------------------------------------------------------------------
// Kernel 1: x_proj = inputs(32768 x 128) x W_ih_pre^T(128 x 512) + b_ih_pre
// ---------------------------------------------------------------------------
__global__ void xproj_kernel(const float* __restrict__ inputs,
                             const unsigned short* __restrict__ wih_p,
                             const float* __restrict__ bih,
                             float* __restrict__ xproj) {
    int gwave = (blockIdx.x * blockDim.x + threadIdx.x) >> 5;
    int lane  = threadIdx.x & 31;
    int nt = gwave & 31;          // 32 N tiles
    int mt = gwave >> 5;          // 2048 M tiles
    bool hi = lane >= 16;
    int row = mt * 16 + (lane & 15);
    const float* rowptr = inputs + (size_t)row * IN_F;

    v8f acc = {};
#pragma unroll
    for (int kt = 0; kt < 4; ++kt) {
        v16bf a = gather_a_f32(rowptr, kt * 32 + (hi ? 8 : 0));
        v16bf b = load_b(wih_p, nt, kt, 4, lane);
        acc = wmma_bf16(a, b, acc);
    }
    int n = nt * 16 + (lane & 15);
#pragma unroll
    for (int r = 0; r < 8; ++r) {
        int M = r + (hi ? 8 : 0);
        int gr = mt * 16 + M;
        xproj[(size_t)gr * HID + n] = acc[r] + bih[n];
    }
}

// ---------------------------------------------------------------------------
// Kernel 2: pre-RNN scan.  4 blocks x 1024 threads; block bt owns 16 batch
// rows.  h (16x512 bf16) double-buffered in LDS.  x_proj step tiles (32 KB)
// are TDM-prefetched one step ahead into a double-buffered LDS tile, hiding
// the global fetch behind the WMMA work.
// ---------------------------------------------------------------------------
__global__ void __launch_bounds__(1024)
pre_rnn_kernel(const float* __restrict__ xproj,
               const unsigned short* __restrict__ whh_p,
               const float* __restrict__ bhh,
               unsigned short* __restrict__ out_pre) {
    __shared__ alignas(16) unsigned short hbuf[2][16 * HID];   // 32 KB
    __shared__ alignas(16) float          xbuf[2][16 * HID];   // 64 KB

    int t = threadIdx.x, bt = blockIdx.x;
    for (int i = t; i < 16 * HID; i += 1024) hbuf[0][i] = 0;

    int wave = t >> 5, lane = t & 31;
    int nt = wave;
    int m = lane & 15;
    bool hi = lane >= 16;
    int kb0 = hi ? 8 : 0;
    int n = nt * 16 + (lane & 15);

    // Prefetch step 0's x_proj tile.
    if (wave == 0) {
        tdm_load_1d_f32((unsigned)(uintptr_t)&xbuf[0][0],
                        xproj + (size_t)bt * 16 * HID, 16 * HID);
        __builtin_amdgcn_s_wait_tensorcnt(0);
    }
    __syncthreads();

    for (int s = 0; s < S_LEN; ++s) {
        const unsigned short* cur = hbuf[s & 1];
        unsigned short* nxt = hbuf[(s + 1) & 1];

        // Async-prefetch next step's x_proj tile while WMMAs run.
        if (wave == 0 && (s + 1) < S_LEN) {
            tdm_load_1d_f32((unsigned)(uintptr_t)&xbuf[(s + 1) & 1][0],
                            xproj + ((size_t)(s + 1) * BATCH + bt * 16) * HID,
                            16 * HID);
        }

        v8f acc = {};
#pragma unroll 4
        for (int kt = 0; kt < 16; ++kt) {
            __builtin_prefetch(&whh_p[((nt * 16 + ((kt + 1) & 15)) * 32) << 4], 0, 0);
            v16bf a = gather_a_u16(cur + m * HID, kt * 32 + kb0);
            v16bf b = load_b(whh_p, nt, kt, 16, lane);
            acc = wmma_bf16(a, b, acc);
        }
        const float* xrow = &xbuf[s & 1][0];
#pragma unroll
        for (int r = 0; r < 8; ++r) {
            int M = r + (hi ? 8 : 0);
            int bg = bt * 16 + M;
            float v = acc[r] + xrow[M * HID + n] + bhh[n];
            unsigned short hb = f2bf(tanhf(v));
            nxt[M * HID + n] = hb;
            out_pre[((size_t)s * BATCH + bg) * HID + n] = hb;
        }
        if (wave == 0) __builtin_amdgcn_s_wait_tensorcnt(0);
        __syncthreads();
    }
}

// ---------------------------------------------------------------------------
// Kernel 3: attention loop (512 iterations) + fc head.
// 4 blocks x 1024 threads; block bt owns 16 batch rows.
// LDS: h (bf16 16x512), ctx (bf16 16x512), scores (f32 512x16) = 64 KB.
// ---------------------------------------------------------------------------
__global__ void __launch_bounds__(1024)
attention_kernel(const unsigned short* __restrict__ out_pre,
                 const unsigned short* __restrict__ wih_p,
                 const unsigned short* __restrict__ whh_p,
                 const float* __restrict__ bih,
                 const float* __restrict__ bhh,
                 const float* __restrict__ Wfc,
                 const float* __restrict__ bfc,
                 float* __restrict__ out) {
    __shared__ alignas(16) unsigned short h_bf[16 * HID];
    __shared__ alignas(16) unsigned short ctx_bf[16 * HID];
    __shared__ float sc[S_LEN * 16];

    int t = threadIdx.x, bt = blockIdx.x;
    int wave = t >> 5, lane = t & 31;
    for (int i = t; i < 16 * HID; i += 1024) h_bf[i] = 0;
    __syncthreads();

    int nt = wave;
    int m = lane & 15;
    bool hi = lane >= 16;
    int kb0 = hi ? 8 : 0;
    int nn = nt * 16 + (lane & 15);

    int bl_c = t >> 6;               // ctx mapping: 64 threads per batch row
    int hbase = (t & 63) << 3;       // 8 consecutive h per thread

    for (int it = 0; it < HID; ++it) {
        // -------- scores[s, bl] = <out_pre[s,b,:], h[bl,:]> (wave-coop dots)
        for (int d = wave; d < S_LEN * 16; d += 32) {
            int s = d >> 4, bl = d & 15;
            int bg = bt * 16 + bl;
            const unsigned short* row = out_pre + ((size_t)s * BATCH + bg) * HID;
            int h0 = lane * 16;
            float sum = 0.f;
#pragma unroll
            for (int j = 0; j < 16; ++j)
                sum += bf2f(row[h0 + j]) * bf2f(h_bf[bl * HID + h0 + j]);
#pragma unroll
            for (int off = 16; off > 0; off >>= 1)
                sum += __shfl_xor(sum, off, 32);
            if (lane == 0) sc[s * 16 + bl] = sum;
        }
        __syncthreads();

        // -------- softmax over s (one wave per batch row)
        if (wave < 16) {
            int bl = wave;
            float mx = -3.4e38f;
            for (int s = lane; s < S_LEN; s += 32)
                mx = fmaxf(mx, sc[s * 16 + bl]);
#pragma unroll
            for (int off = 16; off > 0; off >>= 1)
                mx = fmaxf(mx, __shfl_xor(mx, off, 32));
            float sum = 0.f;
            for (int s = lane; s < S_LEN; s += 32) {
                float e = __expf(sc[s * 16 + bl] - mx);
                sc[s * 16 + bl] = e;
                sum += e;
            }
#pragma unroll
            for (int off = 16; off > 0; off >>= 1)
                sum += __shfl_xor(sum, off, 32);
            float inv = 1.0f / sum;
            for (int s = lane; s < S_LEN; s += 32)
                sc[s * 16 + bl] *= inv;
        }
        __syncthreads();

        // -------- ctx[bl, h] = sum_s w[s,bl] * out_pre[s,b,h]
        {
            int bg = bt * 16 + bl_c;
            float a8[8] = {0.f};
            for (int s = 0; s < S_LEN; ++s) {
                float w = sc[s * 16 + bl_c];
                const unsigned short* row =
                    out_pre + ((size_t)s * BATCH + bg) * HID + hbase;
#pragma unroll
                for (int j = 0; j < 8; ++j) a8[j] += w * bf2f(row[j]);
            }
#pragma unroll
            for (int j = 0; j < 8; ++j)
                ctx_bf[bl_c * HID + hbase + j] = f2bf(a8[j]);
        }
        __syncthreads();

        // -------- h = tanh(ctx @ W_ih_post^T + h @ W_hh_post^T + biases)
        v8f acc = {};
#pragma unroll 4
        for (int kt = 0; kt < 16; ++kt) {
            __builtin_prefetch(&wih_p[((nt * 16 + ((kt + 1) & 15)) * 32) << 4], 0, 0);
            v16bf a = gather_a_u16(ctx_bf + m * HID, kt * 32 + kb0);
            v16bf b = load_b(wih_p, nt, kt, 16, lane);
            acc = wmma_bf16(a, b, acc);
        }
#pragma unroll 4
        for (int kt = 0; kt < 16; ++kt) {
            v16bf a = gather_a_u16(h_bf + m * HID, kt * 32 + kb0);
            v16bf b = load_b(whh_p, nt, kt, 16, lane);
            acc = wmma_bf16(a, b, acc);
        }
        float hv[8];
#pragma unroll
        for (int r = 0; r < 8; ++r)
            hv[r] = tanhf(acc[r] + bih[nn] + bhh[nn]);
        __syncthreads();   // all reads of h_bf complete before overwrite
#pragma unroll
        for (int r = 0; r < 8; ++r) {
            int M = r + (hi ? 8 : 0);
            h_bf[M * HID + nn] = f2bf(hv[r]);
        }
        __syncthreads();
    }

    // -------- fc head: out[b] = h[b,:] . W_fc + b_fc   (O = 1)
    if (wave < 16) {
        float sum = 0.f;
        for (int h = lane; h < HID; h += 32)
            sum += bf2f(h_bf[wave * HID + h]) * Wfc[h];
#pragma unroll
        for (int off = 16; off > 0; off >>= 1)
            sum += __shfl_xor(sum, off, 32);
        if (lane == 0) out[bt * 16 + wave] = sum + bfc[0];
    }
}

// ---------------------------------------------------------------------------
// Host side
// ---------------------------------------------------------------------------
extern "C" void kernel_launch(void* const* d_in, const int* in_sizes, int n_in,
                              void* d_out, int out_size, void* d_ws, size_t ws_size,
                              hipStream_t stream) {
    const float* inputs    = (const float*)d_in[0];   // (S,B,I)
    const float* W_ih_pre  = (const float*)d_in[1];   // (H,I)
    const float* W_hh_pre  = (const float*)d_in[2];   // (H,H)
    const float* b_ih_pre  = (const float*)d_in[3];
    const float* b_hh_pre  = (const float*)d_in[4];
    const float* W_ih_post = (const float*)d_in[5];   // (H,H)
    const float* W_hh_post = (const float*)d_in[6];   // (H,H)
    const float* b_ih_post = (const float*)d_in[7];
    const float* b_hh_post = (const float*)d_in[8];
    const float* W_fc      = (const float*)d_in[9];   // (O,H) = (1,512)
    const float* b_fc      = (const float*)d_in[10];

    char* ws = (char*)d_ws;
    size_t off = 0;
    float*          xproj      = (float*)(ws + off);          off += (size_t)S_LEN * BATCH * HID * 4;  // 64 MB
    unsigned short* out_pre    = (unsigned short*)(ws + off); off += (size_t)S_LEN * BATCH * HID * 2;  // 32 MB
    unsigned short* wih_pre_p  = (unsigned short*)(ws + off); off += (size_t)IN_F * HID * 2;
    unsigned short* whh_pre_p  = (unsigned short*)(ws + off); off += (size_t)HID * HID * 2;
    unsigned short* wih_post_p = (unsigned short*)(ws + off); off += (size_t)HID * HID * 2;
    unsigned short* whh_post_p = (unsigned short*)(ws + off); off += (size_t)HID * HID * 2;

    // Pack weights into bf16 WMMA B-fragment layout.
    prep_b_kernel<<<(IN_F * HID + 255) / 256, 256, 0, stream>>>(W_ih_pre,  wih_pre_p,  IN_F, HID);
    prep_b_kernel<<<(HID * HID + 255) / 256, 256, 0, stream>>>(W_hh_pre,  whh_pre_p,  HID, HID);
    prep_b_kernel<<<(HID * HID + 255) / 256, 256, 0, stream>>>(W_ih_post, wih_post_p, HID, HID);
    prep_b_kernel<<<(HID * HID + 255) / 256, 256, 0, stream>>>(W_hh_post, whh_post_p, HID, HID);

    // x_proj GEMM: 2048 x 32 tiles of 16x16, 8 waves per 256-thread block.
    xproj_kernel<<<(2048 * 32) / 8, 256, 0, stream>>>(inputs, wih_pre_p, b_ih_pre, xproj);

    // Sequential phases: 4 persistent workgroups (16 batch rows each).
    pre_rnn_kernel<<<4, 1024, 0, stream>>>(xproj, whh_pre_p, b_hh_pre, out_pre);
    attention_kernel<<<4, 1024, 0, stream>>>(out_pre, wih_post_p, whh_post_p,
                                             b_ih_post, b_hh_post, W_fc, b_fc,
                                             (float*)d_out);
}
